// RetinaNet_Head_42649025249430
// MI455X (gfx1250) — matrix operand order repaired and checked by reading, not compiled, not run on hardware
//
#include <hip/hip_runtime.h>
#include <hip/hip_bf16.h>
#include <math.h>

// ---------------- problem constants ----------------
#define HH    100
#define WWID  152
#define NPIX  (HH*WWID)          // 15200
#define NIMG  8
#define MTOT  (NIMG*NPIX)        // 121600  (== 128*950, == 256*475)
#define CIN   256
#define KTOT  (CIN*9)            // 2304

typedef __attribute__((ext_vector_type(16))) _Float16 v16h;
typedef __attribute__((ext_vector_type(8)))  float    v8f;

// =====================================================================
// Weight transform: OIHW fp32 -> [OPAD][KTOT] f16, k = (ky*3+kx)*256 + c
// =====================================================================
__global__ __launch_bounds__(256)
void wtransform_kernel(const float* __restrict__ w, _Float16* __restrict__ wt,
                       int O, int OPAD) {
    int idx = blockIdx.x * 256 + threadIdx.x;
    if (idx >= OPAD * KTOT) return;
    int o = idx / KTOT;
    int k = idx - o * KTOT;
    int r = k >> 8;          // ky*3+kx, 0..8
    int c = k & 255;
    float v = 0.f;
    if (o < O) v = w[(size_t)(o * CIN + c) * 9 + r];   // OIHW
    wt[idx] = (_Float16)v;
}

// =====================================================================
// Feature NCHW fp32 -> NHWC f16  ([m][c], m = img*NPIX + h*W + w)
// =====================================================================
__global__ __launch_bounds__(256)
void to_nhwc_f16_kernel(const float* __restrict__ in, _Float16* __restrict__ out) {
    size_t idx = (size_t)blockIdx.x * 256 + threadIdx.x;   // NCHW linear
    int img = (int)(idx / ((size_t)CIN * NPIX));
    int r   = (int)(idx % ((size_t)CIN * NPIX));
    int c   = r / NPIX;
    int p   = r - c * NPIX;
    out[(size_t)(img * NPIX + p) * CIN + c] = (_Float16)in[idx];
}

// =====================================================================
// Implicit-GEMM conv3x3 via WMMA f16 -> f32.
//   A[M=pixels][K=2304] from NHWC f16 activations (zero-padded borders)
//   B[K][N=Cout] from transformed weights stored [n][k]
// Block: 256 thr = 8 waves in 4(M) x 2(N); block tile 128 x 128;
// wave tile 32 x 64 = 2x4 tiles of v_wmma_f32_16x16x32_f16, K-step 32.
// Double-buffered LDS, one barrier/iter; weights staged with
// GLOBAL_LOAD_ASYNC_TO_LDS_B128 fenced by s_wait_asynccnt.
// =====================================================================
template<int COUT, bool RELU, bool OUT_F32>
__global__ __launch_bounds__(256)
void conv3x3_wmma_kernel(const _Float16* __restrict__ act,   // [MTOT][CIN] f16
                         const _Float16* __restrict__ wt,    // [NPAD][KTOT] f16
                         const float*    __restrict__ bias,  // [COUT]
                         _Float16*       __restrict__ outH,  // f16 out (if !OUT_F32)
                         float*          __restrict__ outF)  // f32 out (if OUT_F32)
{
    __shared__ __align__(16) _Float16 sA[2][128][40];    // [buf][m][k]
    __shared__ __align__(16) _Float16 sB[2][128][40];    // [buf][n][k]

    const int t    = threadIdx.x;
    const int lane = t & 31;
    const int wid  = t >> 5;
    const int wm   = wid >> 1;          // 0..3
    const int wn   = wid & 1;           // 0..1
    const int g    = lane >> 4;         // K-half select
    const int row  = lane & 15;

    const int mBase = blockIdx.x * 128;
    const int nBase = blockIdx.y * 128;

    // staging mapping (A and B identical): rowS = t>>1, 16 contiguous k at (t&1)*16
    const int rowS = t >> 1;            // 0..127
    const int offS = (t & 1) * 16;      // 0 or 16 halfs

    // A row pixel decode (constant over K loop)
    const int m   = mBase + rowS;
    const int img = m / NPIX;
    const int rem = m - img * NPIX;
    const int hh0 = rem / WWID;
    const int ww0 = rem - hh0 * WWID;

    const _Float16* wrow = wt + (size_t)(nBase + rowS) * KTOT;

    // LDS raw byte offsets for async weight staging (addr[31:0] == LDS offset)
    unsigned ldsB0[2], ldsB1[2];
    ldsB0[0] = (unsigned)(uintptr_t)&sB[0][rowS][offS];
    ldsB1[0] = (unsigned)(uintptr_t)&sB[0][rowS][offS + 8];
    ldsB0[1] = (unsigned)(uintptr_t)&sB[1][rowS][offS];
    ldsB1[1] = (unsigned)(uintptr_t)&sB[1][rowS][offS + 8];

    const v8f vzero = {0.f,0.f,0.f,0.f,0.f,0.f,0.f,0.f};
    v8f acc[2][4];
#pragma unroll
    for (int mt = 0; mt < 2; ++mt)
#pragma unroll
        for (int nt = 0; nt < 4; ++nt) acc[mt][nt] = vzero;

    union Frag { uint4 u[2]; v16h v; };

    // ---- helpers ----
    auto issueB = [&](int kb, int b) {
        unsigned long long ga = (unsigned long long)(uintptr_t)(wrow + kb + offS);
        asm volatile(
            "global_load_async_to_lds_b128 %0, %2, off\n\t"
            "global_load_async_to_lds_b128 %1, %3, off"
            :: "v"(ldsB0[b]), "v"(ldsB1[b]), "v"(ga), "v"(ga + 16ull)
            : "memory");
    };
    auto loadA = [&](int kb, uint4& r0, uint4& r1) {
        int k  = kb + offS;
        int r  = k >> 8;            // tap index 0..8 (same for both uint4s)
        int c  = k & 255;
        int hh = hh0 + (r / 3) - 1;
        int ww = ww0 + (r % 3) - 1;
        uint4 z = {0u,0u,0u,0u};
        r0 = z; r1 = z;
        if ((unsigned)hh < (unsigned)HH && (unsigned)ww < (unsigned)WWID) {
            const uint4* p = (const uint4*)(act + (size_t)(img * NPIX + hh * WWID + ww) * CIN + c);
            r0 = p[0];
            r1 = p[1];
        }
    };

    // ---- pipeline prologue ----
    uint4 a0, a1;
    issueB(0, 0);
    loadA(0, a0, a1);

    int buf = 0;
    for (int kb = 0; kb < KTOT; kb += 32) {
        // all of this wave's async weight writes into sB[buf] are done:
        asm volatile("s_wait_asynccnt 0" ::: "memory");
        // commit staged A registers into sA[buf]
        *(uint4*)&sA[buf][rowS][offS]     = a0;
        *(uint4*)&sA[buf][rowS][offS + 8] = a1;
        __syncthreads();   // sA[buf]/sB[buf] visible to all waves; frees buf^1

        // prefetch next K-step into the other buffer / registers
        if (kb + 32 < KTOT) {
            issueB(kb + 32, buf ^ 1);
            loadA(kb + 32, a0, a1);
        }

        // ---- load fragments from LDS ----
        Frag a[2], b[4];
#pragma unroll
        for (int mt = 0; mt < 2; ++mt) {
            const _Float16* pa = &sA[buf][wm * 32 + mt * 16 + row][g * 8];
            a[mt].u[0] = *(const uint4*)pa;         // K = g*8 .. g*8+7
            a[mt].u[1] = *(const uint4*)(pa + 16);  // K = 16+g*8 ..
        }
#pragma unroll
        for (int nt = 0; nt < 4; ++nt) {
            const _Float16* pb = &sB[buf][wn * 64 + nt * 16 + row][g * 8];
            b[nt].u[0] = *(const uint4*)pb;         // K = g*8 .. g*8+7
            b[nt].u[1] = *(const uint4*)(pb + 16);  // K = 16+g*8 ..
        }

        // ---- 8x v_wmma_f32_16x16x32_f16 ----
#pragma unroll
        for (int nt = 0; nt < 4; ++nt)
#pragma unroll
            for (int mt = 0; mt < 2; ++mt)
                acc[mt][nt] = __builtin_amdgcn_wmma_f32_16x16x32_f16(
                    false, a[mt].v, false, b[nt].v,
                    (short)0, acc[mt][nt], false, false);

        buf ^= 1;
    }

    // ---- store: C/D layout => element (M = r2 + 8*g, N = row) ----
#pragma unroll
    for (int mt = 0; mt < 2; ++mt) {
        int mrowBase = mBase + wm * 32 + mt * 16 + g * 8;
#pragma unroll
        for (int nt = 0; nt < 4; ++nt) {
            int cout = nBase + wn * 64 + nt * 16 + row;
            if (cout < COUT) {
                float bv = bias[cout];
#pragma unroll
                for (int r2 = 0; r2 < 8; ++r2) {
                    int mm = mrowBase + r2;
                    float v = acc[mt][nt][r2] + bv;
                    if (RELU) v = fmaxf(v, 0.f);
                    if (OUT_F32) outF[(size_t)mm * COUT + cout] = v;
                    else         outH[(size_t)mm * COUT + cout] = (_Float16)v;
                }
            }
        }
    }
}

// =====================================================================
// Direct conv3x3 with Cout=1 (score / centerness): negligible FLOPs.
// =====================================================================
__global__ __launch_bounds__(256)
void conv3x3_c1_kernel(const _Float16* __restrict__ act,  // [MTOT][CIN] f16
                       const float* __restrict__ w,       // (1,256,3,3) OIHW
                       const float* __restrict__ b,
                       float* __restrict__ out)            // [MTOT]
{
    __shared__ float wl[KTOT];
    for (int i = threadIdx.x; i < KTOT; i += 256) {
        int r = i >> 8, c = i & 255;
        wl[i] = w[c * 9 + r];
    }
    __syncthreads();

    int m = blockIdx.x * 256 + threadIdx.x;   // grid exact: 475 blocks
    int img = m / NPIX;
    int rem = m - img * NPIX;
    int h = rem / WWID;
    int ww0 = rem - h * WWID;

    float acc = b[0];
#pragma unroll
    for (int r = 0; r < 9; ++r) {
        int hh = h + r / 3 - 1;
        int ww = ww0 + r % 3 - 1;
        if ((unsigned)hh < (unsigned)HH && (unsigned)ww < (unsigned)WWID) {
            const _Float16* p = act + (size_t)(img * NPIX + hh * WWID + ww) * CIN;
            for (int c = 0; c < CIN; c += 8) {
                uint4 u = *(const uint4*)(p + c);
                const _Float16* hv = (const _Float16*)&u;
#pragma unroll
                for (int j = 0; j < 8; ++j)
                    acc += (float)hv[j] * wl[r * 256 + c + j];
            }
        }
    }
    out[m] = acc;
}

// =====================================================================
// Per-pixel head: softmax(8), erf-PMF, top-4, 1x1 conv 48->64->1,
// sigmoid gating of the classification score. ACC == 1.0 here.
// =====================================================================
__global__ __launch_bounds__(128)
void head_kernel(const float* __restrict__ pred_reg,   // [MTOT][64] NHWC
                 const float* __restrict__ cls_score,  // [MTOT]
                 const float* __restrict__ w1,         // (64,48)
                 const float* __restrict__ b1,         // (64)
                 const float* __restrict__ w2,         // (64)
                 const float* __restrict__ b2,         // (1)
                 float* __restrict__ pred_cls)          // [MTOT]
{
    __shared__ float sw1[64 * 48];
    __shared__ float sb1[64];
    __shared__ float sw2[64];
    for (int i = threadIdx.x; i < 64 * 48; i += 128) sw1[i] = w1[i];
    for (int i = threadIdx.x; i < 64; i += 128) { sb1[i] = b1[i]; sw2[i] = w2[i]; }
    __syncthreads();

    int m = blockIdx.x * 128 + threadIdx.x;   // grid exact: 950 blocks
    const float* pr = pred_reg + (size_t)m * 64;

    float stat[48];
    const float SQRT1_2 = 0.70710678118654752440f;

#pragma unroll 1
    for (int gk = 0; gk < 4; ++gk) {
        float logit[8], lstd[8], prob[8], inv[8], pmf[8];
#pragma unroll
        for (int j = 0; j < 8; ++j) {
            logit[j] = pr[gk * 16 + j];
            lstd[j]  = pr[gk * 16 + 8 + j];
        }
        float mx = logit[0];
#pragma unroll
        for (int j = 1; j < 8; ++j) mx = fmaxf(mx, logit[j]);
        float s = 0.f;
#pragma unroll
        for (int j = 0; j < 8; ++j) { prob[j] = expf(logit[j] - mx); s += prob[j]; }
        float is = 1.f / s;
#pragma unroll
        for (int j = 0; j < 8; ++j) {
            prob[j] *= is;
            inv[j] = SQRT1_2 * expf(-lstd[j]);    // 1/(sigma*sqrt(2))
        }
#pragma unroll 1
        for (int j = 0; j < 8; ++j) {
            float p = 0.f;
#pragma unroll
            for (int i = 0; i < 8; ++i) {
                float diff = (float)(j - i);      // ACC = 1.0
                float hi = 0.5f * (1.f + erff((diff + 1.f) * inv[i]));
                float lo = 0.5f * (1.f + erff((diff - 1.f) * inv[i]));
                p += (hi - lo) * prob[i];
            }
            pmf[j] = p;
        }
        // top-4 (descending, first index wins ties)
        bool used[8] = {false,false,false,false,false,false,false,false};
#pragma unroll
        for (int tt = 0; tt < 4; ++tt) {
            int bi = 0; float bv = -1e30f;
#pragma unroll
            for (int j = 0; j < 8; ++j)
                if (!used[j] && pmf[j] > bv) { bv = pmf[j]; bi = j; }
            used[bi] = true;
            stat[gk * 4 + tt]      = lstd[bi];
            stat[16 + gk * 4 + tt] = prob[bi];
            stat[32 + gk * 4 + tt] = bv;
        }
    }

    // conf: 48 -> 64 (relu) -> 1
    float q = b2[0];
#pragma unroll 1
    for (int o = 0; o < 64; ++o) {
        float hsum = sb1[o];
        const float* wrow = &sw1[o * 48];
#pragma unroll
        for (int i = 0; i < 48; ++i) hsum += wrow[i] * stat[i];
        q += sw2[o] * fmaxf(hsum, 0.f);
    }
    float quality = 1.f / (1.f + expf(-q));
    float cs = cls_score[m];
    pred_cls[m] = quality / (1.f + expf(-cs));
}

// =====================================================================
// Host-side orchestration
// =====================================================================
extern "C" void kernel_launch(void* const* d_in, const int* in_sizes, int n_in,
                              void* d_out, int out_size, void* d_ws, size_t ws_size,
                              hipStream_t stream) {
    const float* feature = (const float*)d_in[0];
    const float* cls_w   = (const float*)d_in[1];
    const float* cls_b   = (const float*)d_in[2];
    const float* box_w   = (const float*)d_in[3];
    const float* box_b   = (const float*)d_in[4];
    const float* score_w = (const float*)d_in[5];
    const float* score_b = (const float*)d_in[6];
    const float* pred_w  = (const float*)d_in[7];
    const float* pred_b  = (const float*)d_in[8];
    const float* ctn_w   = (const float*)d_in[9];
    const float* ctn_b   = (const float*)d_in[10];
    const float* conf_w1 = (const float*)d_in[11];
    const float* conf_b1 = (const float*)d_in[12];
    const float* conf_w2 = (const float*)d_in[13];
    const float* conf_b2 = (const float*)d_in[14];

    // workspace layout (bytes)
    const size_t SZ_ACT   = (size_t)MTOT * CIN * sizeof(_Float16);   // 62,259,200
    const size_t SZ_WT256 = (size_t)256 * KTOT * sizeof(_Float16);   // 1,179,648
    const size_t SZ_WT128 = (size_t)128 * KTOT * sizeof(_Float16);   //   589,824
    char* ws = (char*)d_ws;
    _Float16* actF    = (_Float16*)(ws);
    _Float16* actA    = (_Float16*)(ws + SZ_ACT);
    _Float16* actB    = (_Float16*)(ws + 2 * SZ_ACT);
    _Float16* wt_cls  = (_Float16*)(ws + 3 * SZ_ACT);
    _Float16* wt_box  = (_Float16*)(ws + 3 * SZ_ACT + 4 * SZ_WT256);
    _Float16* wt_pred = (_Float16*)(ws + 3 * SZ_ACT + 8 * SZ_WT256);
    float*    clsbuf  = (float*)   (ws + 3 * SZ_ACT + 8 * SZ_WT256 + SZ_WT128);

    float* out_cls = (float*)d_out;                              // [MTOT]
    float* out_reg = (float*)d_out + MTOT;                       // [MTOT][64]
    float* out_ctn = (float*)d_out + MTOT + (size_t)MTOT * 64;   // [MTOT]

    // ---- weight transforms ----
    const int gW256 = (256 * KTOT + 255) / 256;
    const int gW128 = (128 * KTOT + 255) / 256;
    for (int i = 0; i < 4; ++i) {
        wtransform_kernel<<<gW256, 256, 0, stream>>>(
            cls_w + (size_t)i * 256 * 256 * 9, wt_cls + (size_t)i * 256 * KTOT, 256, 256);
        wtransform_kernel<<<gW256, 256, 0, stream>>>(
            box_w + (size_t)i * 256 * 256 * 9, wt_box + (size_t)i * 256 * KTOT, 256, 256);
    }
    wtransform_kernel<<<gW128, 256, 0, stream>>>(pred_w, wt_pred, 64, 128);

    // ---- feature -> NHWC f16 ----
    to_nhwc_f16_kernel<<<MTOT, 256, 0, stream>>>(feature, actF);

    dim3 gConv(MTOT / 128, 2);     // 950 x 2
    dim3 gPred(MTOT / 128, 1);

    // ---- cls tower (4x conv+relu) ----
    conv3x3_wmma_kernel<256, true, false><<<gConv, 256, 0, stream>>>(
        actF, wt_cls + 0 * (size_t)256 * KTOT, cls_b + 0,   actA, nullptr);
    conv3x3_wmma_kernel<256, true, false><<<gConv, 256, 0, stream>>>(
        actA, wt_cls + 1 * (size_t)256 * KTOT, cls_b + 256, actB, nullptr);
    conv3x3_wmma_kernel<256, true, false><<<gConv, 256, 0, stream>>>(
        actB, wt_cls + 2 * (size_t)256 * KTOT, cls_b + 512, actA, nullptr);
    conv3x3_wmma_kernel<256, true, false><<<gConv, 256, 0, stream>>>(
        actA, wt_cls + 3 * (size_t)256 * KTOT, cls_b + 768, actB, nullptr);
    // cls_score (Cout=1) from cls tower output
    conv3x3_c1_kernel<<<MTOT / 256, 256, 0, stream>>>(actB, score_w, score_b, clsbuf);

    // ---- box tower (4x conv+relu), reuse ping-pong ----
    conv3x3_wmma_kernel<256, true, false><<<gConv, 256, 0, stream>>>(
        actF, wt_box + 0 * (size_t)256 * KTOT, box_b + 0,   actA, nullptr);
    conv3x3_wmma_kernel<256, true, false><<<gConv, 256, 0, stream>>>(
        actA, wt_box + 1 * (size_t)256 * KTOT, box_b + 256, actB, nullptr);
    conv3x3_wmma_kernel<256, true, false><<<gConv, 256, 0, stream>>>(
        actB, wt_box + 2 * (size_t)256 * KTOT, box_b + 512, actA, nullptr);
    conv3x3_wmma_kernel<256, true, false><<<gConv, 256, 0, stream>>>(
        actA, wt_box + 3 * (size_t)256 * KTOT, box_b + 768, actB, nullptr);

    // bbox_pred (Cout=64, no relu) -> directly into pred_reg (NHWC fp32 == output layout)
    conv3x3_wmma_kernel<64, false, true><<<gPred, 256, 0, stream>>>(
        actB, wt_pred, pred_b, nullptr, out_reg);
    // centerness (Cout=1) -> directly into pred_ctn region
    conv3x3_c1_kernel<<<MTOT / 256, 256, 0, stream>>>(actB, ctn_w, ctn_b, out_ctn);

    // ---- distribution stats + quality gating -> pred_cls ----
    head_kernel<<<MTOT / 128, 128, 0, stream>>>(
        out_reg, clsbuf, conf_w1, conf_b1, conf_w2, conf_b2, out_cls);
}
